// Part_Graph_75419625717916
// MI455X (gfx1250) — matrix operand order repaired and checked by reading, not compiled, not run on hardware
//
#include <hip/hip_runtime.h>
#include <hip/hip_bf16.h>
#include <math.h>

// ---------------------------------------------------------------------------
// Part-graph module for MI455X (gfx1250, wave32, WMMA).
// GEMM-shaped reductions (offset convs + deformable main convs) run on
// v_wmma_f32_16x16x32_bf16. A-tiles are staged in LDS in (m,k) order so the
// k->(channel,tap) decode is done once per K-column instead of per element.
// ---------------------------------------------------------------------------

#define HD    10
#define BB    4
#define Himg  128
#define Wimg  128
#define HW    (Himg * Wimg)          // 16384
#define NPIX  (BB * HW)              // 65536
#define EPSV  1e-5f

typedef __attribute__((ext_vector_type(16))) __bf16 v16bf;
typedef __attribute__((ext_vector_type(8)))  float  v8f;

struct Groups {                     // virtual channel-concat: up to 5 groups of 10 ch
    const float* p[5];
    int bstride[5];                 // per-group batch stride (elements)
    int ng;
};

__device__ __forceinline__ v8f wmma_bf16(v16bf a, v16bf b, v8f c) {
    return __builtin_amdgcn_wmma_f32_16x16x32_bf16(false, a, false, b,
                                                   (short)0, c, false, false);
}

// ---------------------------------------------------------------------------
// Pack an [N,K] fp32 weight matrix into WMMA B-fragment order (bf16):
// layout [ntile(2)][kchunk][lane(32)][j(16)], value = W[n, k] with
//   n = ntile*16 + (lane&15),  k = kc*32 + j + 16*(lane>>4),  zero-padded.
// ---------------------------------------------------------------------------
__global__ void pack_b_kernel(const float* __restrict__ w, int N, int K,
                              __bf16* __restrict__ out, int nChunks) {
    int idx = blockIdx.x * blockDim.x + threadIdx.x;
    int total = 2 * nChunks * 32 * 16;
    if (idx >= total) return;
    int j    = idx & 15;
    int lane = (idx >> 4) & 31;
    int kc   = (idx >> 9) % nChunks;
    int nt   = idx / (nChunks * 512);
    int n = nt * 16 + (lane & 15);
    int k = kc * 32 + j + 16 * (lane >> 4);
    float v = (n < N && k < K) ? w[(size_t)n * K + k] : 0.0f;
    out[idx] = (__bf16)v;
}

// ---------------------------------------------------------------------------
// 3x3 conv as im2col GEMM on WMMA. One wave == one 16-pixel row tile.
// The 3x18xC halo patch is staged once in LDS (bf16), then each K-chunk's
// 16x32 A-tile is assembled in LDS and consumed by two WMMAs (N=27 padded).
// ---------------------------------------------------------------------------
template <int C>
__global__ void __launch_bounds__(128)
conv3x3_wmma_kernel(Groups g, const __bf16* __restrict__ packedB,
                    const float* __restrict__ bias, int Nout,
                    float* __restrict__ out) {
    constexpr int KTOT = C * 9;
    constexpr int NCH  = (KTOT + 31) / 32;
    constexpr int PSZ  = C * 54;            // 3 rows x 18 cols x C channels

    __shared__ __bf16 patch[4][PSZ];
    __shared__ __bf16 atile[4][512];

    int lane = threadIdx.x & 31;
    int wave = threadIdx.x >> 5;
    int tile = blockIdx.x * 4 + wave;       // 4096 tiles: b(4) x y(128) x xt(8)
    int b  = tile >> 10;
    int y  = (tile >> 3) & 127;
    int x0 = (tile & 7) << 4;
    int half = lane >> 4;
    int m    = lane & 15;

#pragma unroll
    for (int gi = 0; gi < C / 10; ++gi)
        __builtin_prefetch(g.p[gi] + (size_t)b * g.bstride[gi] + y * Wimg + x0, 0, 0);

    // ---- stage halo patch ----
    for (int e = lane; e < PSZ; e += 32) {
        int c   = e / 54;
        int r   = (e - c * 54) / 18;
        int col = e - c * 54 - r * 18;
        int gy = y + r - 1, gx = x0 + col - 1;
        int gi = c / 10, cc = c - gi * 10;
        float v = 0.0f;
        if (gy >= 0 && gy < Himg && gx >= 0 && gx < Wimg)
            v = g.p[gi][(size_t)b * g.bstride[gi] + cc * HW + gy * Wimg + gx];
        patch[wave][e] = (__bf16)v;
    }
    __syncthreads();

    v8f acc0 = {}; v8f acc1 = {};
    for (int kc = 0; kc < NCH; ++kc) {
        // each lane produces one K-column (kw == lane) of the 16x32 A-tile:
        // k->(c,tap) decoded once, 16 contiguous patch reads follow.
        int kk = kc * 32 + lane;
        __bf16 col[16];
        if (kk < KTOT) {
            int c   = kk / 9;
            int tap = kk - c * 9;
            int base = c * 54 + (tap / 3) * 18 + (tap % 3);
#pragma unroll
            for (int t = 0; t < 16; ++t) col[t] = patch[wave][base + t];
        } else {
#pragma unroll
            for (int t = 0; t < 16; ++t) col[t] = (__bf16)0.0f;
        }
#pragma unroll
        for (int t = 0; t < 16; ++t) atile[wave][t * 32 + lane] = col[t];
        __syncthreads();

        v16bf a;
#pragma unroll
        for (int j = 0; j < 16; ++j) {
            int kw = j + 16 * half;   // contiguous k = 16*half + j, same order as packed B
            a[j] = atile[wave][m * 32 + kw];
        }
        v16bf b0 = *(const v16bf*)(packedB + (size_t)(kc * 32 + lane) * 16);
        acc0 = wmma_bf16(a, b0, acc0);
        v16bf b1 = *(const v16bf*)(packedB + (size_t)((NCH + kc) * 32 + lane) * 16);
        acc1 = wmma_bf16(a, b1, acc1);
        __syncthreads();
    }

    int n = lane & 15;
#pragma unroll
    for (int v = 0; v < 8; ++v) {
        int mr  = v + 8 * half;                 // ISA C/D layout
        int pix = y * Wimg + x0 + mr;
        out[((size_t)b * Nout + n) * HW + pix] = acc0[v] + bias[n];
        int n1 = 16 + n;
        if (n1 < Nout)
            out[((size_t)b * Nout + n1) * HW + pix] = acc1[v] + bias[n1];
    }
}

// ---------------------------------------------------------------------------
// Modulated deformable conv main GEMM. Per 16-pixel tile: cache the 16x9
// bilinear corner weights (mask-premultiplied, zeroed OOB) + clamped indices
// in LDS (shared across all C channels), then reduce K = C*9 on WMMA with the
// same one-decode-per-K-column LDS A-tile staging.
// ---------------------------------------------------------------------------
template <int C, int COUT, int NT>
__global__ void __launch_bounds__(128)
deform_wmma_kernel(Groups g, const float* __restrict__ om,
                   const __bf16* __restrict__ packedB,
                   float* __restrict__ out) {
    constexpr int KTOT = C * 9;
    constexpr int NCH  = (KTOT + 31) / 32;

    __shared__ float  sw[4][144][4];
    __shared__ int    si[4][144][4];
    __shared__ __bf16 atile[4][512];

    int lane = threadIdx.x & 31;
    int wave = threadIdx.x >> 5;
    int tile = blockIdx.x * 4 + wave;
    int b  = tile >> 10;
    int y  = (tile >> 3) & 127;
    int x0 = (tile & 7) << 4;
    int half = lane >> 4;
    int m    = lane & 15;

    const float* omB = om + (size_t)b * 27 * HW;
    __builtin_prefetch(omB + y * Wimg + x0, 0, 0);

    for (int t = lane; t < 144; t += 32) {
        int mm  = t / 9;
        int tap = t - mm * 9;
        int x   = x0 + mm;
        int pix = y * Wimg + x;
        float dy  = omB[(2 * tap) * HW + pix];
        float dx  = omB[(2 * tap + 1) * HW + pix];
        float msk = 1.0f / (1.0f + __expf(-omB[(18 + tap) * HW + pix]));
        float py = (float)y + (float)(tap / 3 - 1) + dy;
        float px = (float)x + (float)(tap % 3 - 1) + dx;
        float fy = floorf(py), fx = floorf(px);
        float wy = py - fy,    wx = px - fx;
        int iy0 = (int)fy, ix0 = (int)fx;
        int iy1 = iy0 + 1, ix1 = ix0 + 1;
        float vy0 = (iy0 >= 0 && iy0 < Himg) ? 1.0f : 0.0f;
        float vy1 = (iy1 >= 0 && iy1 < Himg) ? 1.0f : 0.0f;
        float vx0 = (ix0 >= 0 && ix0 < Wimg) ? 1.0f : 0.0f;
        float vx1 = (ix1 >= 0 && ix1 < Wimg) ? 1.0f : 0.0f;
        sw[wave][t][0] = (1.0f - wy) * (1.0f - wx) * msk * vy0 * vx0;
        sw[wave][t][1] = (1.0f - wy) * wx          * msk * vy0 * vx1;
        sw[wave][t][2] = wy          * (1.0f - wx) * msk * vy1 * vx0;
        sw[wave][t][3] = wy          * wx          * msk * vy1 * vx1;
        int cy0 = min(max(iy0, 0), Himg - 1), cy1 = min(max(iy1, 0), Himg - 1);
        int cx0 = min(max(ix0, 0), Wimg - 1), cx1 = min(max(ix1, 0), Wimg - 1);
        si[wave][t][0] = cy0 * Wimg + cx0;
        si[wave][t][1] = cy0 * Wimg + cx1;
        si[wave][t][2] = cy1 * Wimg + cx0;
        si[wave][t][3] = cy1 * Wimg + cx1;
    }
    __syncthreads();

    v8f acc0 = {}; v8f acc1 = {};
    for (int kc = 0; kc < NCH; ++kc) {
        int kk = kc * 32 + lane;
        __bf16 col[16];
        if (kk < KTOT) {
            int c   = kk / 9;                      // once per chunk per lane
            int tap = kk - c * 9;
            int gi  = c / 10, cc = c - gi * 10;
            const float* xc = g.p[gi] + (size_t)b * g.bstride[gi] + cc * HW;
#pragma unroll
            for (int t = 0; t < 16; ++t) {
                int e = t * 9 + tap;
                float v = sw[wave][e][0] * xc[si[wave][e][0]]
                        + sw[wave][e][1] * xc[si[wave][e][1]]
                        + sw[wave][e][2] * xc[si[wave][e][2]]
                        + sw[wave][e][3] * xc[si[wave][e][3]];
                col[t] = (__bf16)v;
            }
        } else {
#pragma unroll
            for (int t = 0; t < 16; ++t) col[t] = (__bf16)0.0f;
        }
#pragma unroll
        for (int t = 0; t < 16; ++t) atile[wave][t * 32 + lane] = col[t];
        __syncthreads();

        v16bf a;
#pragma unroll
        for (int j = 0; j < 16; ++j) {
            int kw = j + 16 * half;
            a[j] = atile[wave][m * 32 + kw];
        }
        v16bf b0 = *(const v16bf*)(packedB + (size_t)(kc * 32 + lane) * 16);
        acc0 = wmma_bf16(a, b0, acc0);
        if (NT > 1) {
            v16bf b1 = *(const v16bf*)(packedB + (size_t)((NCH + kc) * 32 + lane) * 16);
            acc1 = wmma_bf16(a, b1, acc1);
        }
        __syncthreads();
    }

    int n = lane & 15;
#pragma unroll
    for (int v = 0; v < 8; ++v) {
        int mr  = v + 8 * half;
        int pix = y * Wimg + x0 + mr;
        if (n < COUT)
            out[((size_t)b * COUT + n) * HW + pix] = acc0[v];
        int n1 = 16 + n;
        if (NT > 1 && n1 < COUT)
            out[((size_t)b * COUT + n1) * HW + pix] = acc1[v];
    }
}

// ---------------------------------------------------------------------------
// BatchNorm statistics: sums[c] += sum(x[:,c]), sums[C+c] += sum(x^2).
// ---------------------------------------------------------------------------
__global__ void bn_stats_kernel(const float* __restrict__ x, int C,
                                float* __restrict__ sums) {
    int c = blockIdx.y;
    float a = 0.0f, a2 = 0.0f;
    for (int i = blockIdx.x * blockDim.x + threadIdx.x; i < NPIX;
         i += gridDim.x * blockDim.x) {
        int b = i >> 14, hw = i & (HW - 1);
        float v = x[((size_t)b * C + c) * HW + hw];
        a += v; a2 += v * v;
    }
    __shared__ float s1[256], s2[256];
    s1[threadIdx.x] = a; s2[threadIdx.x] = a2;
    __syncthreads();
    for (int off = 128; off > 0; off >>= 1) {
        if (threadIdx.x < off) {
            s1[threadIdx.x] += s1[threadIdx.x + off];
            s2[threadIdx.x] += s2[threadIdx.x + off];
        }
        __syncthreads();
    }
    if (threadIdx.x == 0) {
        atomicAdd(&sums[c],     s1[0]);
        atomicAdd(&sums[C + c], s2[0]);
    }
}

__global__ void bn_apply_relu_kernel(const float* __restrict__ x,
                                     const float* __restrict__ sums,
                                     const float* __restrict__ gg,
                                     const float* __restrict__ bb,
                                     int C, float* __restrict__ out) {
    int idx = blockIdx.x * blockDim.x + threadIdx.x;
    if (idx >= BB * C * HW) return;
    int c = (idx / HW) % C;
    float inv  = 1.0f / (float)NPIX;
    float mean = sums[c] * inv;
    float var  = sums[C + c] * inv - mean * mean;
    float v = (x[idx] - mean) * rsqrtf(var + EPSV) * gg[c] + bb[c];
    out[idx] = v > 0.0f ? v : 0.0f;
}

// ---------------------------------------------------------------------------
// decomp: t = W1 @ concat(a,b)  (K=20, N=20 per pixel)
// ---------------------------------------------------------------------------
__global__ void decomp_pass1_kernel(const float* __restrict__ a,
                                    const float* __restrict__ bsrc,
                                    const float* __restrict__ W1,
                                    float* __restrict__ t) {
    int p = blockIdx.x * blockDim.x + threadIdx.x;
    if (p >= NPIX) return;
    int b = p >> 14, hw = p & (HW - 1);
    float cat[20];
#pragma unroll
    for (int c = 0; c < 10; ++c) cat[c]      = a[((size_t)b * 10 + c) * HW + hw];
#pragma unroll
    for (int c = 0; c < 10; ++c) cat[10 + c] = bsrc[((size_t)b * 10 + c) * HW + hw];
    for (int o = 0; o < 20; ++o) {
        float s = 0.0f;
#pragma unroll
        for (int c = 0; c < 20; ++c) s += W1[o * 20 + c] * cat[c];
        t[((size_t)b * 20 + o) * HW + hw] = s;
    }
}

__global__ void decomp_pass2_kernel(const float* __restrict__ t,
                                    const float* __restrict__ sums,
                                    const float* __restrict__ g1,
                                    const float* __restrict__ b1,
                                    const float* __restrict__ W2,
                                    const float* __restrict__ b2,
                                    float* __restrict__ out) {
    int p = blockIdx.x * blockDim.x + threadIdx.x;
    if (p >= NPIX) return;
    int b = p >> 14, hw = p & (HW - 1);
    float inv = 1.0f / (float)NPIX;
    float s = b2[0];
    for (int c = 0; c < 20; ++c) {
        float mean = sums[c] * inv;
        float var  = sums[20 + c] * inv - mean * mean;
        float v = (t[((size_t)b * 20 + c) * HW + hw] - mean) * rsqrtf(var + EPSV)
                  * g1[c] + b1[c];
        v = v >= 0.0f ? v : 0.01f * v;              // LeakyReLU(0.01)
        s += W2[c] * v;
    }
    out[(size_t)b * HW + hw] = 1.0f / (1.0f + __expf(-s));
}

__global__ void residual_kernel(const float* __restrict__ xp,
                                const float* __restrict__ t,
                                const float* __restrict__ gamma,
                                float* __restrict__ out) {
    int idx = blockIdx.x * blockDim.x + threadIdx.x;
    if (idx >= BB * HD * HW) return;
    float v = gamma[0] * xp[idx] + t[idx];
    out[idx] = v > 0.0f ? v : 0.0f;
}

// ---------------------------------------------------------------------------
extern "C" void kernel_launch(void* const* d_in, const int* in_sizes, int n_in,
                              void* d_out, int out_size, void* d_ws, size_t ws_size,
                              hipStream_t stream) {
    (void)in_sizes; (void)n_in; (void)out_size; (void)ws_size;
    const float* xf     = (const float*)d_in[0];
    const float* xh     = (const float*)d_in[1];
    const float* xp     = (const float*)d_in[2];
    const float* dfp_w1 = (const float*)d_in[3];
    const float* dfp_g1 = (const float*)d_in[4];
    const float* dfp_b1 = (const float*)d_in[5];
    const float* dfp_w2 = (const float*)d_in[6];
    const float* dfp_b2 = (const float*)d_in[7];
    const float* dhp_w1 = (const float*)d_in[8];
    const float* dhp_g1 = (const float*)d_in[9];
    const float* dhp_b1 = (const float*)d_in[10];
    const float* dhp_w2 = (const float*)d_in[11];
    const float* dhp_b2 = (const float*)d_in[12];
    const float* off1_w = (const float*)d_in[13];
    const float* off1_b = (const float*)d_in[14];
    const float* w1     = (const float*)d_in[15];
    const float* g1     = (const float*)d_in[16];
    const float* b1     = (const float*)d_in[17];
    const float* off2_w = (const float*)d_in[18];
    const float* off2_b = (const float*)d_in[19];
    const float* w2     = (const float*)d_in[20];
    const float* g2     = (const float*)d_in[21];
    const float* b2     = (const float*)d_in[22];
    const float* gamma  = (const float*)d_in[23];

    float* out     = (float*)d_out;
    float* out_xp  = out;                                   // [6,B,10,H,W]
    float* out_afp = out + (size_t)6 * BB * HD * HW;        // [6,B,1,H,W]
    float* out_ahp = out_afp + (size_t)6 * BB * HW;

    // workspace carve (floats)
    float*  ws    = (float*)d_ws;
    float*  tdec  = ws;                                     // B*20*HW
    float*  om    = tdec + (size_t)BB * 20 * HW;            // B*27*HW
    float*  ybuf  = om   + (size_t)BB * 27 * HW;            // B*20*HW
    float*  t1    = ybuf + (size_t)BB * 20 * HW;            // B*20*HW
    float*  sums  = t1   + (size_t)BB * 20 * HW;            // 64
    __bf16* packed = (__bf16*)(sums + 64);                  // 2*15*32*16 bf16 max

    static const int NBRS[6][2] = {{1,5},{0,2},{1,3},{2,4},{3,5},{0,4}};
    const int PIXB = (NPIX + 255) / 256;

    for (int i = 0; i < 6; ++i) {
        const float* xpi  = xp + (size_t)i * BB * HD * HW;
        const float* hsel = xh + (size_t)((i <= 3) ? 0 : 1) * BB * HD * HW;

        // ---- decomp fp ----
        hipMemsetAsync(sums, 0, 64 * sizeof(float), stream);
        decomp_pass1_kernel<<<PIXB, 256, 0, stream>>>(xf, xpi, dfp_w1 + i * 400, tdec);
        bn_stats_kernel<<<dim3(8, 20), 256, 0, stream>>>(tdec, 20, sums);
        decomp_pass2_kernel<<<PIXB, 256, 0, stream>>>(
            tdec, sums, dfp_g1 + i * 20, dfp_b1 + i * 20, dfp_w2 + i * 20,
            dfp_b2 + i, out_afp + (size_t)i * BB * HW);

        // ---- decomp hp ----
        hipMemsetAsync(sums, 0, 64 * sizeof(float), stream);
        decomp_pass1_kernel<<<PIXB, 256, 0, stream>>>(hsel, xpi, dhp_w1 + i * 400, tdec);
        bn_stats_kernel<<<dim3(8, 20), 256, 0, stream>>>(tdec, 20, sums);
        decomp_pass2_kernel<<<PIXB, 256, 0, stream>>>(
            tdec, sums, dhp_g1 + i * 20, dhp_b1 + i * 20, dhp_w2 + i * 20,
            dhp_b2 + i, out_ahp + (size_t)i * BB * HW);

        // ---- deformable block 1: cat(50ch) -> 20ch ----
        Groups gc;
        gc.ng = 5;
        gc.p[0] = xpi;  gc.p[1] = xf;  gc.p[2] = hsel;
        gc.p[3] = xp + (size_t)NBRS[i][0] * BB * HD * HW;
        gc.p[4] = xp + (size_t)NBRS[i][1] * BB * HD * HW;
        for (int q = 0; q < 5; ++q) gc.bstride[q] = HD * HW;

        const int K1 = 50 * 9, nC1 = (K1 + 31) / 32;   // 15 chunks
        pack_b_kernel<<<(2 * nC1 * 512 + 255) / 256, 256, 0, stream>>>(
            off1_w + (size_t)i * 27 * K1, 27, K1, packed, nC1);
        conv3x3_wmma_kernel<50><<<1024, 128, 0, stream>>>(
            gc, packed, off1_b + i * 27, 27, om);
        pack_b_kernel<<<(2 * nC1 * 512 + 255) / 256, 256, 0, stream>>>(
            w1 + (size_t)i * 20 * K1, 20, K1, packed, nC1);
        deform_wmma_kernel<50, 20, 2><<<1024, 128, 0, stream>>>(
            gc, om, packed, ybuf);
        hipMemsetAsync(sums, 0, 64 * sizeof(float), stream);
        bn_stats_kernel<<<dim3(8, 20), 256, 0, stream>>>(ybuf, 20, sums);
        bn_apply_relu_kernel<<<(BB * 20 * HW + 255) / 256, 256, 0, stream>>>(
            ybuf, sums, g1 + i * 20, b1 + i * 20, 20, t1);

        // ---- deformable block 2: 20ch -> 10ch ----
        Groups gd;
        gd.ng = 2;
        gd.p[0] = t1;  gd.p[1] = t1 + (size_t)HD * HW;
        gd.p[2] = t1;  gd.p[3] = t1;  gd.p[4] = t1;
        for (int q = 0; q < 5; ++q) gd.bstride[q] = 20 * HW;

        const int K2 = 20 * 9, nC2 = (K2 + 31) / 32;   // 6 chunks
        pack_b_kernel<<<(2 * nC2 * 512 + 255) / 256, 256, 0, stream>>>(
            off2_w + (size_t)i * 27 * K2, 27, K2, packed, nC2);
        conv3x3_wmma_kernel<20><<<1024, 128, 0, stream>>>(
            gd, packed, off2_b + i * 27, 27, om);
        pack_b_kernel<<<(2 * nC2 * 512 + 255) / 256, 256, 0, stream>>>(
            w2 + (size_t)i * 10 * K2, 10, K2, packed, nC2);
        deform_wmma_kernel<20, 10, 1><<<1024, 128, 0, stream>>>(
            gd, om, packed, ybuf);
        hipMemsetAsync(sums, 0, 64 * sizeof(float), stream);
        bn_stats_kernel<<<dim3(8, 10), 256, 0, stream>>>(ybuf, 10, sums);
        bn_apply_relu_kernel<<<(BB * 10 * HW + 255) / 256, 256, 0, stream>>>(
            ybuf, sums, g2 + i * 10, b2 + i * 10, 10, ybuf);

        // ---- residual ----
        residual_kernel<<<(BB * HD * HW + 255) / 256, 256, 0, stream>>>(
            xpi, ybuf, gamma + i, out_xp + (size_t)i * BB * HD * HW);
    }
}
